// FirstLayer_6511170421226
// MI455X (gfx1250) — compile-verified, matrix operand order checked by reading, not run on hardware
//
#include <hip/hip_runtime.h>
#include <hip/hip_bf16.h>

typedef __attribute__((ext_vector_type(16))) _Float16 v16h;
typedef __attribute__((ext_vector_type(8)))  float    v8f;

#define T_LEN 512
#define B_LEN 512
#define I_LEN 100
#define H_LEN 96
#define G_LEN 288            // 3*H
#define XS    134            // padded row stride (halfs) for lds_x / lds_h: 67 words, conflict-free
#define NWAVES 18            // 288 / 16 N-tiles
#define NTHREADS (NWAVES * 32)

// K index (even base of a half-pair) inside a 16-bit WMMA A/B fragment.
// Per CDNA5 ISA 7.12.2: VGPR v(0..3): K = 2v (+8 for lanes 16..31); VGPR 4..7: +16.
__device__ __forceinline__ int frag_kbase(int p, int hi) {
    return ((p & 4) << 2) + ((p & 3) << 1) + (hi << 3);
}

// Unconditional fragment load (matrix is zero-padded in K where needed).
__device__ __forceinline__ v16h load_frag(const _Float16* base, int stride,
                                          int row, int k0, int hi) {
    union { v16h h; unsigned int u[8]; } f;
    const _Float16* r = base + row * stride;
#pragma unroll
    for (int p = 0; p < 8; ++p)
        f.u[p] = *(const unsigned int*)(r + k0 + frag_kbase(p, hi));
    return f.h;
}

// Masked variant (used only once, outside the time loop, for the K=100 weights).
__device__ __forceinline__ v16h load_frag_m(const _Float16* base, int stride,
                                            int row, int k0, int kmax, int hi) {
    union { v16h h; unsigned int u[8]; } f;
    const _Float16* r = base + row * stride;
#pragma unroll
    for (int p = 0; p < 8; ++p) {
        int k = k0 + frag_kbase(p, hi);
        f.u[p] = (k < kmax) ? *(const unsigned int*)(r + k) : 0u;
    }
    return f.h;
}

__device__ __forceinline__ float sigmoidf_(float x) {
    return 1.0f / (1.0f + __expf(-x));
}

__global__ void __launch_bounds__(NTHREADS)
bigru_fused_kernel(const float* __restrict__ x,
                   const float* __restrict__ wih_f, const float* __restrict__ whh_f,
                   const float* __restrict__ bih_f, const float* __restrict__ bhh_f,
                   const float* __restrict__ wih_b, const float* __restrict__ whh_b,
                   const float* __restrict__ bih_b, const float* __restrict__ bhh_b,
                   float* __restrict__ out) {
    __shared__ __align__(16) _Float16 lds_wih[G_LEN][I_LEN];   // 57600 B
    __shared__ __align__(16) _Float16 lds_whh[G_LEN][H_LEN+2]; // 56448 B
    __shared__ __align__(16) _Float16 lds_x[16][XS];           //  4288 B (cols 100..133 stay 0)
    __shared__ __align__(16) _Float16 lds_h[16][XS];           //  4288 B (cols 96..133 stay 0)
    __shared__ __align__(16) float    lds_xp[16][292];         // 18688 B (8*292 % 64 == 32)
    __shared__ __align__(16) float    lds_hp[16][292];         // 18688 B

    const int tid  = threadIdx.x;
    const int lane = tid & 31;
    const int wid  = tid >> 5;          // 0..17 -> N-tile
    const int hi   = (lane >> 4) & 1;
    const int ln   = lane & 15;
    const int dir  = blockIdx.x >> 5;   // 0 = forward, 1 = backward
    const int b0   = (blockIdx.x & 31) * 16;

    const float* wih = dir ? wih_b : wih_f;
    const float* whh = dir ? whh_b : whh_f;
    const float* bih = dir ? bih_b : bih_f;
    const float* bhh = dir ? bhh_b : bhh_f;

    // ---- one-time staging: weights f32->f16, h0 = 0, x-pad = 0 ----
    for (int i = tid; i < G_LEN * I_LEN; i += NTHREADS)
        lds_wih[i / I_LEN][i % I_LEN] = (_Float16)wih[i];
    for (int i = tid; i < G_LEN * H_LEN; i += NTHREADS)
        lds_whh[i / H_LEN][i % H_LEN] = (_Float16)whh[i];
    for (int i = tid; i < 16 * XS; i += NTHREADS) {
        (&lds_h[0][0])[i] = (_Float16)0.0f;
        (&lds_x[0][0])[i] = (_Float16)0.0f;
    }

    // x[t, b0:b0+16, :] is 1600 contiguous floats; each thread owns <=3 of them.
    const float* xblk = x + (size_t)b0 * I_LEN;      // + t*51200 per step
    const int i0 = tid, i1 = tid + NTHREADS, i2 = tid + 2 * NTHREADS;
    const int o0 = (i0 / I_LEN) * XS + (i0 % I_LEN); // flat lds_x offsets
    const int o1 = (i1 / I_LEN) * XS + (i1 % I_LEN);
    const int o2 = (i2 / I_LEN) * XS + (i2 % I_LEN);
    const bool has2 = (i2 < 16 * I_LEN);

    // prologue: prefetch x for the first step into registers
    int t0 = dir ? (T_LEN - 1) : 0;
    float xr0 = xblk[(size_t)t0 * B_LEN * I_LEN + i0];
    float xr1 = xblk[(size_t)t0 * B_LEN * I_LEN + i1];
    float xr2 = has2 ? xblk[(size_t)t0 * B_LEN * I_LEN + i2] : 0.0f;

    __syncthreads();

    // ---- B fragments (weights^T) held in registers for all 512 steps ----
    const int ncol = wid * 16 + ln;
    v16h Bih[4], Bhh[3];
#pragma unroll
    for (int c = 0; c < 4; ++c)
        Bih[c] = load_frag_m(&lds_wih[0][0], I_LEN, ncol, c * 32, I_LEN, hi);
#pragma unroll
    for (int c = 0; c < 3; ++c)
        Bhh[c] = load_frag(&lds_whh[0][0], H_LEN + 2, ncol, c * 32, hi);

    const float bias_i = bih[ncol];
    const float bias_h = bhh[ncol];

    // ---- sequential time loop ----
    for (int s = 0; s < T_LEN; ++s) {
        const int t = dir ? (T_LEN - 1 - s) : s;

        // commit prefetched x (f32->f16) to LDS
        _Float16* lx = &lds_x[0][0];
        lx[o0] = (_Float16)xr0;
        lx[o1] = (_Float16)xr1;
        if (has2) lx[o2] = (_Float16)xr2;
        __syncthreads();

        // kick off next step's x loads; latency hides behind WMMA + gate math
        if (s + 1 < T_LEN) {
            const int tn = dir ? (T_LEN - 2 - s) : (s + 1);
            const float* xg = xblk + (size_t)tn * B_LEN * I_LEN;
            xr0 = xg[i0];
            xr1 = xg[i1];
            if (has2) xr2 = xg[i2];
        }

        // xp-tile = x @ Wih^T + b_ih   (K padded to 128, unconditional frags)
        {
            v8f acc = {0.f, 0.f, 0.f, 0.f, 0.f, 0.f, 0.f, 0.f};
#pragma unroll
            for (int c = 0; c < 4; ++c) {
                v16h a = load_frag(&lds_x[0][0], XS, ln, c * 32, hi);
                acc = __builtin_amdgcn_wmma_f32_16x16x32_f16(
                    false, a, false, Bih[c], (short)0, acc, false, false);
            }
#pragma unroll
            for (int v = 0; v < 8; ++v)
                lds_xp[v + 8 * hi][ncol] = acc[v] + bias_i;
        }
        // hp-tile = h @ Whh^T + b_hh   (K = 96 -> 3 K32 chunks)
        {
            v8f acc = {0.f, 0.f, 0.f, 0.f, 0.f, 0.f, 0.f, 0.f};
#pragma unroll
            for (int c = 0; c < 3; ++c) {
                v16h a = load_frag(&lds_h[0][0], XS, ln, c * 32, hi);
                acc = __builtin_amdgcn_wmma_f32_16x16x32_f16(
                    false, a, false, Bhh[c], (short)0, acc, false, false);
            }
#pragma unroll
            for (int v = 0; v < 8; ++v)
                lds_hp[v + 8 * hi][ncol] = acc[v] + bias_h;
        }
        __syncthreads();

        // gate math (fp32), update h (f16 for next GEMM), stream result to HBM
        for (int i = tid; i < 16 * H_LEN; i += NTHREADS) {
            int m = i / H_LEN, j = i % H_LEN;
            float r = sigmoidf_(lds_xp[m][j]           + lds_hp[m][j]);
            float z = sigmoidf_(lds_xp[m][H_LEN + j]   + lds_hp[m][H_LEN + j]);
            float n = tanhf(lds_xp[m][2 * H_LEN + j] + r * lds_hp[m][2 * H_LEN + j]);
            float hprev = (float)lds_h[m][j];
            float hnew  = (1.0f - z) * n + z * hprev;
            lds_h[m][j] = (_Float16)hnew;
            out[((size_t)t * B_LEN + b0 + m) * (2 * H_LEN) + dir * H_LEN + j] = hnew;
        }
        // next iteration's post-stage barrier orders lds_h/lds_x writes vs GEMM reads
    }
}

extern "C" void kernel_launch(void* const* d_in, const int* in_sizes, int n_in,
                              void* d_out, int out_size, void* d_ws, size_t ws_size,
                              hipStream_t stream) {
    (void)in_sizes; (void)n_in; (void)d_ws; (void)ws_size; (void)out_size;
    const float* x     = (const float*)d_in[0];
    const float* wih_f = (const float*)d_in[1];
    const float* whh_f = (const float*)d_in[2];
    const float* bih_f = (const float*)d_in[3];
    const float* bhh_f = (const float*)d_in[4];
    const float* wih_b = (const float*)d_in[5];
    const float* whh_b = (const float*)d_in[6];
    const float* bih_b = (const float*)d_in[7];
    const float* bhh_b = (const float*)d_in[8];
    float* out = (float*)d_out;

    // 32 batch tiles x 2 directions; each block owns its slice's full recurrence.
    bigru_fused_kernel<<<dim3(64), dim3(NTHREADS), 0, stream>>>(
        x, wih_f, whh_f, bih_f, bhh_f, wih_b, whh_b, bih_b, bhh_b, out);
}
